// FeatureTransMIL_188978561046
// MI455X (gfx1250) — matrix operand order repaired and checked by reading, not compiled, not run on hardware
//
#include <hip/hip_runtime.h>
#include <hip/hip_bf16.h>
#include <utility>

typedef __attribute__((ext_vector_type(16))) _Float16 v16h;
typedef __attribute__((ext_vector_type(8)))  _Float16 v8h;
typedef __attribute__((ext_vector_type(8)))  float    v8f;
typedef __attribute__((ext_vector_type(2)))  float    v2f;

#if defined(__has_builtin)
#if __has_builtin(__builtin_amdgcn_wmma_f32_16x16x4_f32)
#define HAVE_WMMA_F32X4 1
#endif
#endif
#ifndef HAVE_WMMA_F32X4
#define HAVE_WMMA_F32X4 0
#endif

// ---------------- problem constants ----------------
#define BB    2
#define NTOK  4096
#define NSEQ  4097      /* +cls */
#define NP    4352      /* padded to multiple of 256 landmarks */
#define PADF  255       /* zeros padded at FRONT */
#define CC    512
#define INDIM 1024
#define HN    8
#define DH    64
#define NL    256       /* landmarks */
#define LGRP  17        /* tokens per landmark (4352/256) */
#define QKVN  1536

// ---------------- fragment loaders ----------------
// A fragment (16x32 f16): lane L elem e -> M=L&15, K=e+(L>>4)*16
__device__ __forceinline__ v16h load_a16(const _Float16* pa) {
    v8h lo = *(const v8h*)(pa);
    v8h hi = *(const v8h*)(pa + 8);
    v16h a;
#pragma unroll
    for (int e = 0; e < 8; ++e) { a[e] = lo[e]; a[8 + e] = hi[e]; }
    return a;
}
// B fragment (32x16 f16): lane L elem e -> N=L&15, K=e+(L>>4)*16
template <int TRANSB>
__device__ __forceinline__ v16h load_b16(const _Float16* pb, int ldb) {
    v16h b;
    if (TRANSB) {
        v8h lo = *(const v8h*)(pb);
        v8h hi = *(const v8h*)(pb + 8);
#pragma unroll
        for (int e = 0; e < 8; ++e) { b[e] = lo[e]; b[8 + e] = hi[e]; }
    } else {
#pragma unroll
        for (int e = 0; e < 16; ++e) b[e] = pb[(long long)e * ldb];
    }
    return b;
}

// =====================================================================
// f16 WMMA GEMM: C[M,N] = act(scale * A @ op(B) + bias)
// A: f16 row-major MxK (lda). TRANSB=1: B is NxK row-major (C=A@B^T),
// contiguous fragment loads. TRANSB=0: B is KxN (strided loads).
// One wave computes a 16x64 tile: one A fragment feeds 4 back-to-back
// v_wmma issues. Software-pipelined: k+1 fragments are loaded before the
// k-step wmmas so loads overlap matrix ops. grid=(N/64, M/16, batch).
// M mult 16, N mult 64, K mult 32 (guaranteed by problem dims).
// =====================================================================
template <int TRANSB>
__global__ void gemm_f16_wmma(const _Float16* __restrict__ A,
                              const _Float16* __restrict__ B,
                              float* __restrict__ Cf,
                              _Float16* __restrict__ Ch,
                              const float* __restrict__ bias,
                              int K, int lda, int ldb, int ldc,
                              long long sA, long long sB, long long sC,
                              float scale, int relu)
{
    const int lane = threadIdx.x;           // 0..31, wave32
    const int l15  = lane & 15;
    const int hi16 = lane >> 4;             // 0 or 1
    const int tn   = blockIdx.x * 64;
    const int tm   = blockIdx.y * 16;
    A += (long long)blockIdx.z * sA;
    B += (long long)blockIdx.z * sB;
    const long long cbase = (long long)blockIdx.z * sC;
    const int rowA = tm + l15;

    const _Float16* pa = A + (long long)rowA * lda + hi16 * 16;
    const _Float16* pb[4];
#pragma unroll
    for (int j = 0; j < 4; ++j) {
        if (TRANSB) pb[j] = B + (long long)(tn + j * 16 + l15) * ldb + hi16 * 16;
        else        pb[j] = B + (long long)(hi16 * 16) * ldb + tn + j * 16 + l15;
    }

    v8f acc[4] = {};
    // prologue: fragments for k0 = 0
    v16h a_c = load_a16(pa);
    v16h b_c[4];
#pragma unroll
    for (int j = 0; j < 4; ++j) b_c[j] = load_b16<TRANSB>(pb[j], ldb);

    for (int k0 = 32; k0 < K; k0 += 32) {
        pa += 32;
#pragma unroll
        for (int j = 0; j < 4; ++j) pb[j] += TRANSB ? 32 : (long long)32 * ldb;
        __builtin_prefetch((const void*)(pa + 32), 0, 1);
        if (TRANSB) __builtin_prefetch((const void*)(pb[0] + 32), 0, 1);
        // issue next-step loads first so they overlap this step's wmmas
        v16h a_n = load_a16(pa);
        v16h b_n[4];
#pragma unroll
        for (int j = 0; j < 4; ++j) b_n[j] = load_b16<TRANSB>(pb[j], ldb);
#pragma unroll
        for (int j = 0; j < 4; ++j)
            acc[j] = __builtin_amdgcn_wmma_f32_16x16x32_f16(
                         false, a_c, false, b_c[j], (short)0, acc[j], false, false);
        a_c = a_n;
#pragma unroll
        for (int j = 0; j < 4; ++j) b_c[j] = b_n[j];
    }
#pragma unroll
    for (int j = 0; j < 4; ++j)
        acc[j] = __builtin_amdgcn_wmma_f32_16x16x32_f16(
                     false, a_c, false, b_c[j], (short)0, acc[j], false, false);

#pragma unroll
    for (int j = 0; j < 4; ++j) {
        const int colB = tn + j * 16 + l15;
        const float bcol = bias ? bias[colB] : 0.0f;
#pragma unroll
        for (int r = 0; r < 8; ++r) {
            float v = acc[j][r] * scale + bcol;
            if (relu) v = fmaxf(v, 0.0f);
            const int row = tm + hi16 * 8 + r;              // C/D layout
            const long long off = cbase + (long long)row * ldc + colB;
            if (Cf) Cf[off] = v;
            if (Ch) Ch[off] = (_Float16)v;
        }
    }
}

// =====================================================================
// f32 WMMA GEMM (V_WMMA_F32_16X16X4_F32) for Moore-Penrose iterations.
// C = scale * A @ B, 256x256x256, batch 16, 16x64 tile, pipelined.
// =====================================================================
__global__ void gemm_f32_wmma(const float* __restrict__ A,
                              const float* __restrict__ B,
                              float* __restrict__ C,
                              int K, int lda, int ldb, int ldc,
                              long long sA, long long sB, long long sC,
                              float scale)
{
    const int lane = threadIdx.x;
    const int l15  = lane & 15;
    const int hi16 = lane >> 4;
    const int tn   = blockIdx.x * 64;
    const int tm   = blockIdx.y * 16;
    A += (long long)blockIdx.z * sA;
    B += (long long)blockIdx.z * sB;
    const long long cbase = (long long)blockIdx.z * sC;
    const int rowA = tm + l15;

    v8f acc[4] = {};
#if HAVE_WMMA_F32X4
    // A 16x4 f32: VGPR r -> K = k0 + hi16*2 + r (ISA 32-bit A 16x4)
    const float* pa = A + (long long)rowA * lda + hi16 * 2;
    // B 4x16 f32: VGPR r -> K = k0 + hi16*2 + r, N striped over lanes
    const float* pb = B + (long long)(hi16 * 2) * ldb + tn + l15;

    v2f a_c; a_c[0] = pa[0]; a_c[1] = pa[1];
    v2f b_c[4];
#pragma unroll
    for (int j = 0; j < 4; ++j) { b_c[j][0] = pb[j * 16]; b_c[j][1] = pb[ldb + j * 16]; }

    for (int k0 = 4; k0 < K; k0 += 4) {
        pa += 4;
        pb += (long long)4 * ldb;
        v2f a_n; a_n[0] = pa[0]; a_n[1] = pa[1];
        v2f b_n[4];
#pragma unroll
        for (int j = 0; j < 4; ++j) { b_n[j][0] = pb[j * 16]; b_n[j][1] = pb[ldb + j * 16]; }
#pragma unroll
        for (int j = 0; j < 4; ++j)
            acc[j] = __builtin_amdgcn_wmma_f32_16x16x4_f32(
                         false, a_c, false, b_c[j], (short)0, acc[j], false, false);
        a_c = a_n;
#pragma unroll
        for (int j = 0; j < 4; ++j) b_c[j] = b_n[j];
    }
#pragma unroll
    for (int j = 0; j < 4; ++j)
        acc[j] = __builtin_amdgcn_wmma_f32_16x16x4_f32(
                     false, a_c, false, b_c[j], (short)0, acc[j], false, false);
#else
#pragma unroll
    for (int j = 0; j < 4; ++j)
        for (int r = 0; r < 8; ++r) {
            const int row = tm + hi16 * 8 + r;
            const int col = tn + j * 16 + l15;
            float s = 0.0f;
            for (int k = 0; k < K; ++k)
                s += A[(long long)row * lda + k] * B[(long long)k * ldb + col];
            acc[j][r] = s;
        }
#endif
#pragma unroll
    for (int j = 0; j < 4; ++j)
        for (int r = 0; r < 8; ++r) {
            const int row = tm + hi16 * 8 + r;
            C[cbase + (long long)row * ldc + tn + j * 16 + l15] = acc[j][r] * scale;
        }
}

// ---------------- block reductions (blockDim == 256) ----------------
__device__ __forceinline__ float bred_sum(float v, float* sm) {
    const int t = threadIdx.x;
    __syncthreads(); sm[t] = v; __syncthreads();
    for (int s = 128; s > 0; s >>= 1) { if (t < s) sm[t] += sm[t + s]; __syncthreads(); }
    float r = sm[0]; __syncthreads(); return r;
}
__device__ __forceinline__ float bred_max(float v, float* sm) {
    const int t = threadIdx.x;
    __syncthreads(); sm[t] = v; __syncthreads();
    for (int s = 128; s > 0; s >>= 1) { if (t < s) sm[t] = fmaxf(sm[t], sm[t + s]); __syncthreads(); }
    float r = sm[0]; __syncthreads(); return r;
}

// ---------------- elementwise / reshuffle kernels ----------------
__global__ void k_convert_f16(const float* __restrict__ in, _Float16* __restrict__ out, long long n) {
    long long i = (long long)blockIdx.x * 256 + threadIdx.x;
    if (i < n) out[i] = (_Float16)in[i];
}
__global__ void k_zero_f16(_Float16* __restrict__ p, long long n) {
    long long i = (long long)blockIdx.x * 256 + threadIdx.x;
    if (i < n) p[i] = (_Float16)0.0f;
}
// out[N x K] = (f16) in[K x N]^T   (weights pre-transposed once per call)
__global__ void k_transpose_f16(const float* __restrict__ in, _Float16* __restrict__ out, int K, int N) {
    long long i = (long long)blockIdx.x * 256 + threadIdx.x;
    if (i < (long long)K * N) {
        int k = (int)(i / N), n = (int)(i % N);
        out[(long long)n * K + k] = (_Float16)in[i];
    }
}
// h(B,4097,512): row0 = cls, rows 1.. = relu(fc1) output
__global__ void k_assemble(const float* __restrict__ fc1out, const float* __restrict__ cls,
                           float* __restrict__ h) {
    long long i = (long long)blockIdx.x * 256 + threadIdx.x;
    const long long n = (long long)BB * NSEQ * CC;
    if (i >= n) return;
    int c = (int)(i & (CC - 1));
    long long r = i >> 9;
    int t = (int)(r % NSEQ), b = (int)(r / NSEQ);
    h[i] = (t == 0) ? cls[c] : fc1out[((long long)b * NTOK + (t - 1)) * CC + c];
}
// LayerNorm each (b,t) row of h; write f16 into padded buffer at row 255+t
__global__ void k_ln_rows(const float* __restrict__ h, const float* __restrict__ g,
                          const float* __restrict__ be, _Float16* __restrict__ xpad) {
    __shared__ float sm[256];
    const int t = threadIdx.x;
    const long long r = blockIdx.x;                       // 0..B*NSEQ-1
    const float* x = h + r * CC;
    float a0 = x[t], a1 = x[t + 256];
    float mu = bred_sum(a0 + a1, sm) * (1.0f / CC);
    float d0 = a0 - mu, d1 = a1 - mu;
    float var = bred_sum(d0 * d0 + d1 * d1, sm) * (1.0f / CC);
    float rstd = rsqrtf(var + 1e-5f);
    int b = (int)(r / NSEQ), i = (int)(r % NSEQ);
    long long o = ((long long)b * NP + PADF + i) * CC;
    xpad[o + t]       = (_Float16)(d0 * rstd * g[t] + be[t]);
    xpad[o + t + 256] = (_Float16)(d1 * rstd * g[t + 256] + be[t + 256]);
}
// split fused qkv (f16, rows B*NP x 1536) into q (x0.125), k, v in (b,h,i,d)
__global__ void k_split_qkv(const _Float16* __restrict__ qkv,
                            _Float16* __restrict__ qh, _Float16* __restrict__ kh,
                            _Float16* __restrict__ vh) {
    long long i = (long long)blockIdx.x * 256 + threadIdx.x;
    const long long n = (long long)BB * HN * NP * DH;
    if (i >= n) return;
    int d = (int)(i & (DH - 1));
    long long r = i >> 6;
    int tok = (int)(r % NP);
    long long bh = r / NP;
    int hh = (int)(bh & (HN - 1)), b = (int)(bh >> 3);
    long long base = ((long long)b * NP + tok) * QKVN + hh * DH + d;
    qh[i] = (_Float16)((float)qkv[base] * 0.125f);   // SCALE = 64^-0.5
    kh[i] = qkv[base + CC];
    vh[i] = qkv[base + 2 * CC];
}
// landmark means over 17 consecutive tokens
__global__ void k_landmarks(const _Float16* __restrict__ qh, const _Float16* __restrict__ kh,
                            _Float16* __restrict__ qlh, _Float16* __restrict__ klh) {
    const int d = threadIdx.x;                 // 64 threads
    const int g = blockIdx.x % NL;
    const int bh = blockIdx.x / NL;
    long long base = ((long long)bh * NP + g * LGRP) * DH + d;
    float sq = 0.f, sk = 0.f;
    for (int j = 0; j < LGRP; ++j) { sq += (float)qh[base + (long long)j * DH]; sk += (float)kh[base + (long long)j * DH]; }
    long long o = ((long long)bh * NL + g) * DH + d;
    qlh[o] = (_Float16)(sq * (1.0f / LGRP));
    klh[o] = (_Float16)(sk * (1.0f / LGRP));
}
// row softmax, row = blockIdx.x, length len; optional f32 / f16 outputs
__global__ void k_softmax(const float* __restrict__ in, float* __restrict__ outF,
                          _Float16* __restrict__ outH, int len) {
    __shared__ float sm[256];
    const int t = threadIdx.x;
    const long long base = (long long)blockIdx.x * len;
    float m = -3.4e38f;
    for (int i = t; i < len; i += 256) m = fmaxf(m, in[base + i]);
    m = bred_max(m, sm);
    float s = 0.f;
    for (int i = t; i < len; i += 256) s += __expf(in[base + i] - m);
    s = bred_sum(s, sm);
    const float inv = 1.0f / s;
    for (int i = t; i < len; i += 256) {
        float e = __expf(in[base + i] - m) * inv;
        if (outF) outF[base + i] = e;
        if (outH) outH[base + i] = (_Float16)e;
    }
}
__global__ void k_set2(float* s) { s[0] = 0.f; s[1] = 0.f; }
// s[0]=max row-sum(|x|), s[1]=max col-sum(|x|) over all 16 matrices (global)
__global__ void k_pinv_norm(const float* __restrict__ x, float* __restrict__ s) {
    __shared__ float sm[256];
    const int t = threadIdx.x, mat = blockIdx.x;
    const float* xm = x + (long long)mat * NL * NL;
    float rs = 0.f, cs = 0.f;
    for (int c = 0; c < NL; ++c) rs += fabsf(xm[(long long)t * NL + c]);
    for (int r = 0; r < NL; ++r) cs += fabsf(xm[(long long)r * NL + t]);
    float rmax = bred_max(rs, sm);
    float cmax = bred_max(cs, sm);
    if (t == 0) {   // positive floats: int compare == float compare
        atomicMax((int*)&s[0], __float_as_int(rmax));
        atomicMax((int*)&s[1], __float_as_int(cmax));
    }
}
// z = x^T / (s0*s1), batched 16 matrices of 256x256
__global__ void k_pinv_init_z(const float* __restrict__ x, float* __restrict__ z,
                              const float* __restrict__ s) {
    long long i = (long long)blockIdx.x * 256 + threadIdx.x;
    const long long n = 16LL * NL * NL;
    if (i >= n) return;
    int j = (int)(i & 255), r = (int)((i >> 8) & 255);
    long long m = i >> 16;
    z[i] = x[(m << 16) + ((long long)j << 8) + r] * (1.0f / (s[0] * s[1]));
}
// out = alpha*I - in  (in-place safe), batched 16 of 256x256
__global__ void k_aIminus(const float* __restrict__ in, float* __restrict__ out, float alpha) {
    long long i = (long long)blockIdx.x * 256 + threadIdx.x;
    const long long n = 16LL * NL * NL;
    if (i >= n) return;
    int j = (int)(i & 255), r = (int)((i >> 8) & 255);
    out[i] = (r == j ? alpha : 0.0f) - in[i];
}
// depthwise 1-D conv over tokens (kernel 33, cross-correlation), += into outattn
__global__ void k_resconv(const _Float16* __restrict__ vh, const float* __restrict__ rk,
                          float* __restrict__ oa) {
    long long i = (long long)blockIdx.x * 256 + threadIdx.x;
    const long long n = (long long)BB * HN * NP * DH;
    if (i >= n) return;
    int d = (int)(i & (DH - 1));
    long long r = i >> 6;
    int tok = (int)(r % NP);
    long long bh = r / NP;
    int hh = (int)(bh & (HN - 1));
    float acc = 0.f;
    for (int t = 0; t < 33; ++t) {
        int s = tok + t - 16;
        if (s >= 0 && s < NP) acc += (float)vh[(bh * NP + s) * DH + d] * rk[hh * 33 + t];
    }
    oa[i] += acc;
}
// (b,h,i,d) -> (b, i, h*64+d) f16 for the output projection
__global__ void k_merge(const float* __restrict__ oa, _Float16* __restrict__ Oh) {
    long long i = (long long)blockIdx.x * 256 + threadIdx.x;
    const long long n = (long long)BB * NP * CC;
    if (i >= n) return;
    int hd = (int)(i & (CC - 1));
    long long r = i >> 9;
    int tok = (int)(r % NP), b = (int)(r / NP);
    int hh = hd >> 6, d = hd & 63;
    Oh[i] = (_Float16)oa[(((long long)b * HN + hh) * NP + tok) * DH + d];
}
// h += proj[:, -4097:]   (rows 255..4351 of padded projection)
__global__ void k_resid(float* __restrict__ h, const float* __restrict__ proj) {
    long long i = (long long)blockIdx.x * 256 + threadIdx.x;
    const long long n = (long long)BB * NSEQ * CC;
    if (i >= n) return;
    int c = (int)(i & (CC - 1));
    long long r = i >> 9;
    int t = (int)(r % NSEQ), b = (int)(r / NSEQ);
    h[i] += proj[((long long)b * NP + PADF + t) * CC + c];
}
// PPEG: y = f + dw7(f) + dw5(f) + dw3(f) on the 64x64 token grid
__global__ void k_ppeg(const float* __restrict__ h, float* __restrict__ hn,
                       const float* __restrict__ w7, const float* __restrict__ b7,
                       const float* __restrict__ w5, const float* __restrict__ b5,
                       const float* __restrict__ w3, const float* __restrict__ b3) {
    const int c = threadIdx.x;                 // 512 threads
    const int p = blockIdx.x & (NTOK - 1);
    const int b = blockIdx.x / NTOK;
    const int y = p >> 6, x = p & 63;
    const float* hb = h + ((long long)b * NSEQ + 1) * CC;
    float acc = hb[(long long)p * CC + c] + b7[c] + b5[c] + b3[c];
    for (int ky = 0; ky < 7; ++ky) { int yy = y + ky - 3; if ((unsigned)yy >= 64u) continue;
        for (int kx = 0; kx < 7; ++kx) { int xx = x + kx - 3; if ((unsigned)xx >= 64u) continue;
            acc += hb[((long long)(yy * 64 + xx)) * CC + c] * w7[c * 49 + ky * 7 + kx]; } }
    for (int ky = 0; ky < 5; ++ky) { int yy = y + ky - 2; if ((unsigned)yy >= 64u) continue;
        for (int kx = 0; kx < 5; ++kx) { int xx = x + kx - 2; if ((unsigned)xx >= 64u) continue;
            acc += hb[((long long)(yy * 64 + xx)) * CC + c] * w5[c * 25 + ky * 5 + kx]; } }
    for (int ky = 0; ky < 3; ++ky) { int yy = y + ky - 1; if ((unsigned)yy >= 64u) continue;
        for (int kx = 0; kx < 3; ++kx) { int xx = x + kx - 1; if ((unsigned)xx >= 64u) continue;
            acc += hb[((long long)(yy * 64 + xx)) * CC + c] * w3[c * 9 + ky * 3 + kx]; } }
    hn[((long long)b * NSEQ + 1 + p) * CC + c] = acc;
}
__global__ void k_copycls(const float* __restrict__ h, float* __restrict__ hn) {
    hn[(long long)blockIdx.x * NSEQ * CC + threadIdx.x] =
        h[(long long)blockIdx.x * NSEQ * CC + threadIdx.x];
}
// final LayerNorm of token 0 -> d_out (B,512) f32
__global__ void k_lnout(const float* __restrict__ h, const float* __restrict__ g,
                        const float* __restrict__ be, float* __restrict__ out) {
    __shared__ float sm[256];
    const int t = threadIdx.x, b = blockIdx.x;
    const float* x = h + (long long)b * NSEQ * CC;
    float a0 = x[t], a1 = x[t + 256];
    float mu = bred_sum(a0 + a1, sm) * (1.0f / CC);
    float d0 = a0 - mu, d1 = a1 - mu;
    float var = bred_sum(d0 * d0 + d1 * d1, sm) * (1.0f / CC);
    float rstd = rsqrtf(var + 1e-5f);
    out[(long long)b * CC + t]       = d0 * rstd * g[t] + be[t];
    out[(long long)b * CC + t + 256] = d1 * rstd * g[t + 256] + be[t + 256];
}

// =====================================================================
extern "C" void kernel_launch(void* const* d_in, const int* in_sizes, int n_in,
                              void* d_out, int out_size, void* d_ws, size_t ws_size,
                              hipStream_t stream) {
    (void)in_sizes; (void)n_in; (void)out_size; (void)ws_size;
    const float* h_in   = (const float*)d_in[0];
    const float* w_fc1  = (const float*)d_in[1];
    const float* b_fc1  = (const float*)d_in[2];
    const float* cls    = (const float*)d_in[3];
    const float* ln1_g  = (const float*)d_in[4];
    const float* ln1_b  = (const float*)d_in[5];
    const float* qkv1_w = (const float*)d_in[6];
    const float* out1_w = (const float*)d_in[7];
    const float* out1_b = (const float*)d_in[8];
    const float* res1_k = (const float*)d_in[9];
    const float* w7 = (const float*)d_in[10]; const float* b7 = (const float*)d_in[11];
    const float* w5 = (const float*)d_in[12]; const float* b5 = (const float*)d_in[13];
    const float* w3 = (const float*)d_in[14]; const float* b3 = (const float*)d_in[15];
    const float* ln2_g  = (const float*)d_in[16];
    const float* ln2_b  = (const float*)d_in[17];
    const float* qkv2_w = (const float*)d_in[18];
    const float* out2_w = (const float*)d_in[19];
    const float* out2_b = (const float*)d_in[20];
    const float* res2_k = (const float*)d_in[21];
    const float* lnf_g  = (const float*)d_in[22];
    const float* lnf_b  = (const float*)d_in[23];

    // ---------------- workspace arena ----------------
    char* p = (char*)d_ws;
    auto alloc = [&](size_t bytes) -> char* {
        char* r = p; p += (bytes + 255) & ~((size_t)255); return r;
    };
    const long long NE_H    = (long long)BB * NSEQ * CC;
    const long long NE_XPAD = (long long)BB * NP * CC;
    const long long NE_QKV  = (long long)BB * NP * QKVN;
    const long long NE_PH   = (long long)BB * HN * NP * DH;       // per-head q/k/v
    const long long NE_LM   = 16LL * NL * DH;
    const long long NE_A2   = 16LL * NL * NL;
    const long long NE_LG   = 16LL * NP * NL;                     // logits (attn1/attn3)

    _Float16* wfc1T = (_Float16*)alloc((size_t)CC * INDIM * 2);
    _Float16* qkv1T = (_Float16*)alloc((size_t)QKVN * CC * 2);
    _Float16* qkv2T = (_Float16*)alloc((size_t)QKVN * CC * 2);
    _Float16* out1T = (_Float16*)alloc((size_t)CC * CC * 2);
    _Float16* out2T = (_Float16*)alloc((size_t)CC * CC * 2);
    float*    hbuf1 = (float*)alloc((size_t)NE_H * 4);
    float*    hbuf2 = (float*)alloc((size_t)NE_H * 4);
    _Float16* xpadh = (_Float16*)alloc((size_t)NE_XPAD * 2);
    _Float16* qkvh  = (_Float16*)alloc((size_t)NE_QKV * 2);
    _Float16* qh    = (_Float16*)alloc((size_t)NE_PH * 2);
    _Float16* kh    = (_Float16*)alloc((size_t)NE_PH * 2);
    _Float16* vh    = (_Float16*)alloc((size_t)NE_PH * 2);
    _Float16* qlh   = (_Float16*)alloc((size_t)NE_LM * 2);
    _Float16* klh   = (_Float16*)alloc((size_t)NE_LM * 2);
    float*    attn2f= (float*)alloc((size_t)NE_A2 * 4);
    float*    zA    = (float*)alloc((size_t)NE_A2 * 4);
    float*    zB    = (float*)alloc((size_t)NE_A2 * 4);
    float*    xzb   = (float*)alloc((size_t)NE_A2 * 4);
    float*    bufA  = (float*)alloc((size_t)NE_A2 * 4);
    float*    bufB  = (float*)alloc((size_t)NE_A2 * 4);
    _Float16* zh    = (_Float16*)alloc((size_t)NE_A2 * 2);
    _Float16* th    = (_Float16*)alloc((size_t)NE_LM * 2);
    _Float16* uh    = (_Float16*)alloc((size_t)NE_LM * 2);
    float*    scal  = (float*)alloc(256);
    float*    oa    = (float*)alloc((size_t)NE_PH * 4);           // attn out (b,h,i,d)
    _Float16* Oh    = (_Float16*)alloc((size_t)NE_XPAD * 2);
    float*    proj  = (float*)alloc((size_t)NE_XPAD * 4);
    _Float16* a1h   = (_Float16*)alloc((size_t)NE_LG * 2);
    _Float16* a3h   = (_Float16*)alloc((size_t)NE_LG * 2);
    float*    logit = (float*)alloc((size_t)NE_LG * 4);
    // fc1 temporaries live only before attention -> overlap into logits region
    _Float16* hA     = (_Float16*)logit;                               // 16 MB
    float*    fc1out = (float*)((char*)logit + (size_t)16 * 1024 * 1024 + 256);

    auto NB = [](long long n) { return dim3((unsigned)((n + 255) / 256)); };
    auto gemm16 = [&](const _Float16* A, const _Float16* Bm, float* Cf, _Float16* Ch,
                      const float* bias, int M, int N, int K, int lda, int ldb, int ldc,
                      long long sA, long long sB, long long sC, int batch,
                      float scale, int transB, int relu) {
        dim3 g(N / 64, M / 16, batch);
        if (transB)
            gemm_f16_wmma<1><<<g, 32, 0, stream>>>(A, Bm, Cf, Ch, bias, K, lda, ldb, ldc,
                                                   sA, sB, sC, scale, relu);
        else
            gemm_f16_wmma<0><<<g, 32, 0, stream>>>(A, Bm, Cf, Ch, bias, K, lda, ldb, ldc,
                                                   sA, sB, sC, scale, relu);
    };
    auto gemm32 = [&](const float* A, const float* Bm, float* Cm, float scale) {
        dim3 g(NL / 64, NL / 16, 16);
        gemm_f32_wmma<<<g, 32, 0, stream>>>(A, Bm, Cm, NL, NL, NL, NL,
                                            (long long)NL * NL, (long long)NL * NL,
                                            (long long)NL * NL, scale);
    };

    // ---------------- weight prep (f16, transposed to NxK) ----------------
    k_transpose_f16<<<NB((long long)INDIM * CC), 256, 0, stream>>>(w_fc1, wfc1T, INDIM, CC);
    k_transpose_f16<<<NB((long long)CC * QKVN), 256, 0, stream>>>(qkv1_w, qkv1T, CC, QKVN);
    k_transpose_f16<<<NB((long long)CC * QKVN), 256, 0, stream>>>(qkv2_w, qkv2T, CC, QKVN);
    k_transpose_f16<<<NB((long long)CC * CC), 256, 0, stream>>>(out1_w, out1T, CC, CC);
    k_transpose_f16<<<NB((long long)CC * CC), 256, 0, stream>>>(out2_w, out2T, CC, CC);

    // ---------------- fc1 + relu + cls assembly ----------------
    k_convert_f16<<<NB((long long)BB * NTOK * INDIM), 256, 0, stream>>>(h_in, hA, (long long)BB * NTOK * INDIM);
    gemm16(hA, wfc1T, fc1out, nullptr, b_fc1, BB * NTOK, CC, INDIM,
           INDIM, INDIM, CC, 0, 0, 0, 1, 1.0f, 1, 1);
    k_assemble<<<NB(NE_H), 256, 0, stream>>>(fc1out, cls, hbuf1);

    // ---------------- one nystrom attention layer ----------------
    auto attn_layer = [&](float* hbuf, const _Float16* qkvT, const _Float16* outT,
                          const float* outb, const float* resk,
                          const float* lg, const float* lb) {
        k_zero_f16<<<NB(NE_XPAD), 256, 0, stream>>>(xpadh, NE_XPAD);           // front pad rows
        k_ln_rows<<<dim3(BB * NSEQ), 256, 0, stream>>>(hbuf, lg, lb, xpadh);
        // qkv projection (no bias)
        gemm16(xpadh, qkvT, nullptr, qkvh, nullptr, BB * NP, QKVN, CC,
               CC, CC, QKVN, 0, 0, 0, 1, 1.0f, 1, 0);
        k_split_qkv<<<NB(NE_PH), 256, 0, stream>>>(qkvh, qh, kh, vh);
        k_landmarks<<<dim3(16 * NL), 64, 0, stream>>>(qh, kh, qlh, klh);
        // attn1 = softmax(q @ k_l^T)   (b*h batched)
        gemm16(qh, klh, logit, nullptr, nullptr, NP, NL, DH, DH, DH, NL,
               (long long)NP * DH, (long long)NL * DH, (long long)NP * NL, 16, 1.0f, 1, 0);
        k_softmax<<<dim3(16 * NP), 256, 0, stream>>>(logit, nullptr, a1h, NL);
        // attn2 = softmax(q_l @ k_l^T) -> f32 for pinv
        gemm16(qlh, klh, xzb, nullptr, nullptr, NL, NL, DH, DH, DH, NL,
               (long long)NL * DH, (long long)NL * DH, (long long)NL * NL, 16, 1.0f, 1, 0);
        k_softmax<<<dim3(16 * NL), 256, 0, stream>>>(xzb, attn2f, nullptr, NL);
        // attn3 = softmax(q_l @ k^T)
        gemm16(qlh, kh, logit, nullptr, nullptr, NL, NP, DH, DH, DH, NP,
               (long long)NL * DH, (long long)NP * DH, (long long)NL * NP, 16, 1.0f, 1, 0);
        k_softmax<<<dim3(16 * NL), 256, 0, stream>>>(logit, nullptr, a3h, NP);
        // Moore-Penrose pinv of attn2 (6 Newton-Schulz iterations, f32 WMMA)
        k_set2<<<1, 1, 0, stream>>>(scal);
        k_pinv_norm<<<dim3(16), 256, 0, stream>>>(attn2f, scal);
        k_pinv_init_z<<<NB(NE_A2), 256, 0, stream>>>(attn2f, zA, scal);
        float* zc = zA; float* zn = zB;
        for (int it = 0; it < 6; ++it) {
            gemm32(attn2f, zc, xzb, 1.0f);                       // xz = x @ z
            k_aIminus<<<NB(NE_A2), 256, 0, stream>>>(xzb, bufA, 7.0f);
            gemm32(xzb, bufA, bufB, 1.0f);
            k_aIminus<<<NB(NE_A2), 256, 0, stream>>>(bufB, bufB, 15.0f);
            gemm32(xzb, bufB, bufA, 1.0f);
            k_aIminus<<<NB(NE_A2), 256, 0, stream>>>(bufA, bufA, 13.0f);
            gemm32(zc, bufA, zn, 0.25f);
            std::swap(zc, zn);
        }
        k_convert_f16<<<NB(NE_A2), 256, 0, stream>>>(zc, zh, NE_A2);
        // t = attn3 @ v ; u = pinv @ t ; out = attn1 @ u
        gemm16(a3h, vh, nullptr, th, nullptr, NL, DH, NP, NP, DH, DH,
               (long long)NL * NP, (long long)NP * DH, (long long)NL * DH, 16, 1.0f, 0, 0);
        gemm16(zh, th, nullptr, uh, nullptr, NL, DH, NL, NL, DH, DH,
               (long long)NL * NL, (long long)NL * DH, (long long)NL * DH, 16, 1.0f, 0, 0);
        gemm16(a1h, uh, oa, nullptr, nullptr, NP, DH, NL, NL, DH, DH,
               (long long)NP * NL, (long long)NL * DH, (long long)NP * DH, 16, 1.0f, 0, 0);
        // + depthwise residual conv over tokens
        k_resconv<<<NB(NE_PH), 256, 0, stream>>>(vh, resk, oa);
        // merge heads, project, residual add (last 4097 rows)
        k_merge<<<NB(NE_XPAD), 256, 0, stream>>>(oa, Oh);
        gemm16(Oh, outT, proj, nullptr, outb, BB * NP, CC, CC, CC, CC, CC,
               0, 0, 0, 1, 1.0f, 1, 0);
        k_resid<<<NB(NE_H), 256, 0, stream>>>(hbuf, proj);
    };

    attn_layer(hbuf1, qkv1T, out1T, out1_b, res1_k, ln1_g, ln1_b);
    // PPEG positional conv block
    k_ppeg<<<dim3(BB * NTOK), CC, 0, stream>>>(hbuf1, hbuf2, w7, b7, w5, b5, w3, b3);
    k_copycls<<<dim3(BB), CC, 0, stream>>>(hbuf1, hbuf2);
    attn_layer(hbuf2, qkv2T, out2T, out2_b, res2_k, ln2_g, ln2_b);
    // final LN of cls token
    k_lnout<<<dim3(BB), 256, 0, stream>>>(hbuf2, lnf_g, lnf_b, (float*)d_out);
}
